// WinIntegral_47708496724773
// MI455X (gfx1250) — matrix operand order repaired and checked by reading, not compiled, not run on hardware
//
#include <hip/hip_runtime.h>
#include <cmath>

typedef __bf16 bf16;
typedef __attribute__((ext_vector_type(16))) __bf16 v16bf;
typedef __attribute__((ext_vector_type(8)))  float  v8f;

#define WMMA_BF16(a, b, c) \
  __builtin_amdgcn_wmma_f32_16x16x32_bf16(false, (a), false, (b), (short)0, (c), false, false)

// ---------------------------------------------------------------------------
// Problem constants
// ---------------------------------------------------------------------------
#define DIMC   384
#define HEADS  12
#define HDIM   32          // 384/12
#define WIN    7
#define NTOK   49          // 7*7
#define NPAD   64          // padded token count for WMMA tiling
#define NWIN   1024        // 16 * 8 * 8 windows
#define NCOL   50176       // NWIN * NTOK
#define NPAIR  12288       // NWIN * HEADS
#define IMGHW  56

// ---------------------------------------------------------------------------
// fp32 -> bf16 flat conversion (weights)
// ---------------------------------------------------------------------------
__global__ void __launch_bounds__(256) cvt_f32_bf16(const float* __restrict__ in,
                                                    bf16* __restrict__ out, int n) {
  int i = blockIdx.x * 256 + threadIdx.x;
  if (i < n) out[i] = (bf16)in[i];
}

// ---------------------------------------------------------------------------
// Build XwT[j][c]  (bf16, column-major-per-token: contiguous K for B operand)
// Applies the cyclic shift (roll by -3,-3) and window partition as indexing.
// ---------------------------------------------------------------------------
__global__ void __launch_bounds__(256) build_xwt(const float* __restrict__ x,
                                                 bf16* __restrict__ XwT) {
  int idx = blockIdx.x * 256 + threadIdx.x;          // 50176*384 threads exactly
  int j = idx / DIMC, c = idx - j * DIMC;
  int wi = j / NTOK, n = j - wi * NTOK;
  int b = wi >> 6, g = wi & 63, gy = g >> 3, gx = g & 7;
  int hs = gy * WIN + n / WIN;
  int ws = gx * WIN + (n % WIN);
  int h = hs + 3; if (h >= IMGHW) h -= IMGHW;        // x_shifted[i] = x[(i+3)%56]
  int w = ws + 3; if (w >= IMGHW) w -= IMGHW;
  XwT[idx] = (bf16)x[(((size_t)b * DIMC + c) * IMGHW + h) * IMGHW + w];
}

// ---------------------------------------------------------------------------
// Continuous-position-bias table -> dense bias[head][64][64] (fp32).
// Padded columns get -1e30 so softmax masking is automatic.
// ---------------------------------------------------------------------------
__device__ __forceinline__ float rel_map(int d) {
  float f = (float)(d + (WIN - 1));                  // in [0, 12], non-negative
  float g = (f > 0.0f) ? (log1pf(f) / logf(1.3f)) : 0.0f;
  g += (float)(WIN - 1);
  if (g > (float)(2 * WIN - 2)) g = (float)(2 * WIN - 2);
  if (g < 0.0f) g = 0.0f;
  return g;
}

__global__ void __launch_bounds__(256) build_bias(const float* __restrict__ cpb,
                                                  float* __restrict__ bias) {
  int idx = blockIdx.x * 256 + threadIdx.x;          // 12*64*64 threads
  if (idx >= HEADS * NPAD * NPAD) return;
  int hd = idx >> 12, rem = idx & 4095;
  int row = rem >> 6, col = rem & 63;
  float v;
  if (col >= NTOK) {
    v = -1e30f;                                      // mask padded key tokens
  } else if (row >= NTOK) {
    v = 0.0f;                                        // don't-care rows
  } else {
    int iy1 = row / WIN, ix1 = row % WIN;
    int iy2 = col / WIN, ix2 = col % WIN;
    float r0 = rel_map(iy1 - iy2);
    float r1 = rel_map(ix1 - ix2);
    int ri = (int)(r0 * (float)(2 * WIN - 1) + r1);
    v = cpb[ri * HEADS + hd];
  }
  bias[idx] = v;
}

// ---------------------------------------------------------------------------
// Zero-fill (padding regions of Q/K/V fragment buffers must be 0)
// ---------------------------------------------------------------------------
__global__ void __launch_bounds__(256) zero_f4(float4* __restrict__ p, int n) {
  int i = blockIdx.x * 256 + threadIdx.x;
  if (i < n) p[i] = make_float4(0.f, 0.f, 0.f, 0.f);
}

// ---------------------------------------------------------------------------
// Shared WMMA GEMM mainloop:  C[64 x 128] tile of  A[M x 384] * B(col-major)
// A: row-major bf16, row stride 384.  B: stored as Bt[col][k], k contiguous.
// 8 waves in 2x4 grid, each wave owns a 32x32 tile (2x2 fragment accums).
// LDS tiles are filled with CDNA5 async global->LDS copies (GVS mode: uniform
// 64-bit SGPR base + per-lane 32-bit byte offset, per-lane LDS dest address),
// directly in WMMA fragment order so each lane reads contiguous 32B fragments.
// ---------------------------------------------------------------------------
__device__ __forceinline__ void gemm_mainloop(const bf16* __restrict__ A,
                                              const bf16* __restrict__ Bt,
                                              int row0, int col0,
                                              bf16* Alds, bf16* Blds,
                                              v8f acc[2][2]) {
  const int tid  = threadIdx.x;
  const int lane = tid & 31;
  const int wave = tid >> 5;
  const int wm = wave >> 2;          // 0..1
  const int wn = wave & 3;           // 0..3

  // Workgroup-relative LDS byte addresses (low 32 bits of the generic address
  // of a __shared__ pointer = LDS aperture offset).
  const unsigned aldsb = (unsigned)(unsigned long long)(void*)Alds;
  const unsigned bldsb = (unsigned)(unsigned long long)(void*)Blds;
  const unsigned long long abase = (unsigned long long)A;
  const unsigned long long bbase = (unsigned long long)Bt;

  for (int kk = 0; kk < 12; ++kk) {
    const int k0 = kk << 5;
    // ---- async-stage A tile (64 rows x 32 k) into fragment-order LDS ----
    for (int p = tid; p < 1024; p += 256) {        // 1024 bf16-pairs (b32 each)
      int v = p & 7, ln = (p >> 3) & 31, sub = p >> 8;
      int m = (ln & 15) + (sub << 4);
      int k = ((v >> 2) << 4) + ((ln >> 4) << 3) + ((v & 3) << 1);
      unsigned goff = (unsigned)((((row0 + m) * DIMC) + k0 + k) * 2);
      unsigned loff = aldsb + (unsigned)((((sub << 9) + (ln << 4) + (v << 1))) * 2);
      asm volatile("global_load_async_to_lds_b32 %0, %1, %2"
                   :: "v"(loff), "v"(goff), "s"(abase) : "memory");
    }
    // ---- async-stage B tile (32 k x 128 cols) into fragment-order LDS ----
    for (int p = tid; p < 2048; p += 256) {        // 2048 bf16-pairs (b32 each)
      int v = p & 7, ln = (p >> 3) & 31, subn = p >> 8;
      int col = col0 + (subn << 4) + (ln & 15);
      int k = ((v >> 2) << 4) + ((ln >> 4) << 3) + ((v & 3) << 1);
      unsigned goff = (unsigned)((col * DIMC + k0 + k) * 2);
      unsigned loff = bldsb + (unsigned)((((subn << 9) + (ln << 4) + (v << 1))) * 2);
      asm volatile("global_load_async_to_lds_b32 %0, %1, %2"
                   :: "v"(loff), "v"(goff), "s"(bbase) : "memory");
    }
    asm volatile("s_wait_asynccnt 0" ::: "memory");  // our async fills done
    __syncthreads();                                  // everyone's fills done

    v16bf a0 = *reinterpret_cast<const v16bf*>(&Alds[((wm * 2 + 0) << 9) + (lane << 4)]);
    v16bf a1 = *reinterpret_cast<const v16bf*>(&Alds[((wm * 2 + 1) << 9) + (lane << 4)]);
    v16bf b0 = *reinterpret_cast<const v16bf*>(&Blds[((wn * 2 + 0) << 9) + (lane << 4)]);
    v16bf b1 = *reinterpret_cast<const v16bf*>(&Blds[((wn * 2 + 1) << 9) + (lane << 4)]);

    acc[0][0] = WMMA_BF16(a0, b0, acc[0][0]);
    acc[0][1] = WMMA_BF16(a0, b1, acc[0][1]);
    acc[1][0] = WMMA_BF16(a1, b0, acc[1][0]);
    acc[1][1] = WMMA_BF16(a1, b1, acc[1][1]);
    __syncthreads();                                  // reads done before refill
  }
}

// ---------------------------------------------------------------------------
// GEMM1: QKV = Wqkv[1152x384] * XwT'   -> scatter into WMMA fragment buffers
//   Q: A-fragments (token = M rows), K: B-fragments (token = N cols),
//   V: B-fragments for P*V (token = K rows, head-dim = N cols)
// Each (window,head) owns 2048 bf16 elements (4 fragments) in each buffer.
// ---------------------------------------------------------------------------
__global__ void __launch_bounds__(256) gemm_qkv(const bf16* __restrict__ Wq,
                                                const bf16* __restrict__ XwT,
                                                bf16* __restrict__ Qf,
                                                bf16* __restrict__ Kf,
                                                bf16* __restrict__ Vf) {
  __shared__ __align__(32) bf16 Alds[2048];
  __shared__ __align__(32) bf16 Blds[4096];
  const int row0 = blockIdx.y * 64;
  const int col0 = blockIdx.x * 128;
  const int lane = threadIdx.x & 31;
  const int wave = threadIdx.x >> 5;
  const int wm = wave >> 2, wn = wave & 3;

  v8f acc[2][2] = {{{}, {}}, {{}, {}}};
  gemm_mainloop(Wq, XwT, row0, col0, Alds, Blds, acc);

  const int l15 = lane & 15, half = lane >> 4;
  for (int im = 0; im < 2; ++im)
    for (int in = 0; in < 2; ++in)
      for (int r = 0; r < 8; ++r) {
        int o = row0 + wm * 32 + im * 16 + r + half * 8;   // output channel
        int j = col0 + wn * 32 + in * 16 + l15;            // window*49+token
        bf16 bv = (bf16)acc[im][in][r];
        int part = o / DIMC;                               // 0=q 1=k 2=v
        int oo = o - part * DIMC;
        int hd = oo >> 5, dd = oo & 31;
        int wi = j / NTOK, n = j - wi * NTOK;
        size_t base = (size_t)(wi * HEADS + hd) * 2048;
        if (part == 0) {        // Q: A-fragment, m = token, k = dd
          int hk = (dd >> 3) & 1;
          int e = ((dd >> 4) << 3) | (dd & 7);
          int sub = n >> 4, lp = (n & 15) + (hk << 4);
          Qf[base + (sub << 9) + (lp << 4) + e] = bv;
        } else if (part == 1) { // K: B-fragment, n = token, k = dd
          int hk = (dd >> 3) & 1, e = ((dd >> 4) << 3) | (dd & 7);
          int nt = n >> 4, lp = (n & 15) + (hk << 4);
          Kf[base + (nt << 9) + (lp << 4) + e] = bv;
        } else {                // V: B-fragment, k = token, n = dd
          int kt = n >> 5, kl = n & 31;
          int hk = (kl >> 3) & 1, e = ((kl >> 4) << 3) | (kl & 7);
          int nt2 = dd >> 4, lp = (dd & 15) + (hk << 4);
          Vf[base + ((kt * 2 + nt2) << 9) + (lp << 4) + e] = bv;
        }
      }
}

// ---------------------------------------------------------------------------
// Attention: one wave per (window, head).
//   S = q*k^T (4x4 WMMAs), + scale + CPB bias, softmax (shfl reductions),
//   P staged to LDS in A-fragment order, Y = P*V (2x2 WMMAs).
// ---------------------------------------------------------------------------
__global__ void __launch_bounds__(256) attn_kernel(const bf16* __restrict__ Qf,
                                                   const bf16* __restrict__ Kf,
                                                   const bf16* __restrict__ Vf,
                                                   const float* __restrict__ bias,
                                                   bf16* __restrict__ YT) {
  __shared__ __align__(32) bf16 Plds[8 * 1024];
  const int lane = threadIdx.x & 31;
  const int wave = threadIdx.x >> 5;
  const int pair = blockIdx.x * 8 + wave;            // < 12288
  const int wi = pair / HEADS, hd = pair - wi * HEADS;

  const bf16* qb = Qf + (size_t)pair * 2048;
  const bf16* kb = Kf + (size_t)pair * 2048;
  const bf16* vb = Vf + (size_t)pair * 2048;
  bf16* pw = Plds + wave * 1024;

  v16bf kf[4], vf[4];
#pragma unroll
  for (int t = 0; t < 4; ++t)
    kf[t] = *reinterpret_cast<const v16bf*>(kb + (t << 9) + (lane << 4));
#pragma unroll
  for (int t = 0; t < 4; ++t)
    vf[t] = *reinterpret_cast<const v16bf*>(vb + (t << 9) + (lane << 4));

  const float scale = 0.17677669529663687f;          // 1/sqrt(32)
  const int l15 = lane & 15, half = lane >> 4;
  const float* bh = bias + hd * NPAD * NPAD;

  for (int mt = 0; mt < 4; ++mt) {
    v16bf aq = *reinterpret_cast<const v16bf*>(qb + (mt << 9) + (lane << 4));
    v8f sacc[4];
#pragma unroll
    for (int nt = 0; nt < 4; ++nt) {
      v8f z = {};
      sacc[nt] = WMMA_BF16(aq, kf[nt], z);
    }
    // logits = S*scale + bias  (bias = -1e30 in padded cols -> auto mask)
    float lg[4][8];
#pragma unroll
    for (int nt = 0; nt < 4; ++nt)
#pragma unroll
      for (int r = 0; r < 8; ++r) {
        int row = mt * 16 + r + half * 8;
        int col = nt * 16 + l15;
        lg[nt][r] = sacc[nt][r] * scale + bh[row * NPAD + col];
      }
    // softmax over the 49 (padded 64) key tokens; each row lives across the
    // 16 lanes of one half-wave and the 4 nt registers.
#pragma unroll
    for (int r = 0; r < 8; ++r) {
      float mx = lg[0][r];
      mx = fmaxf(mx, lg[1][r]); mx = fmaxf(mx, lg[2][r]); mx = fmaxf(mx, lg[3][r]);
      for (int off = 8; off; off >>= 1) mx = fmaxf(mx, __shfl_xor(mx, off, 32));
      float sum = 0.0f;
#pragma unroll
      for (int nt = 0; nt < 4; ++nt) {
        float e = __expf(lg[nt][r] - mx);
        lg[nt][r] = e; sum += e;
      }
      for (int off = 8; off; off >>= 1) sum += __shfl_xor(sum, off, 32);
      float inv = 1.0f / sum;
#pragma unroll
      for (int nt = 0; nt < 4; ++nt) lg[nt][r] *= inv;
    }
    // write P (bf16) to LDS directly in A-fragment order
#pragma unroll
    for (int nt = 0; nt < 4; ++nt)
#pragma unroll
      for (int r = 0; r < 8; ++r) {
        int rr  = r + half * 8;              // fragment M (0..15)
        int col = nt * 16 + l15;             // K (0..63)
        int kt = col >> 5, kl = col & 31;
        int hk = (kl >> 3) & 1;
        int e = ((kl >> 4) << 3) | (kl & 7);
        pw[(kt << 9) + (((rr & 15) + (hk << 4)) << 4) + e] = (bf16)lg[nt][r];
      }
    asm volatile("s_wait_dscnt 0" ::: "memory");     // intra-wave LDS RAW

    v8f yacc[2] = {{}, {}};
#pragma unroll
    for (int kt = 0; kt < 2; ++kt) {
      v16bf pf = *reinterpret_cast<const v16bf*>(pw + (kt << 9) + (lane << 4));
      yacc[0] = WMMA_BF16(pf, vf[kt * 2 + 0], yacc[0]);
      yacc[1] = WMMA_BF16(pf, vf[kt * 2 + 1], yacc[1]);
    }
    // store Y rows (skip padded tokens); YT[j][c], c contiguous
#pragma unroll
    for (int nt2 = 0; nt2 < 2; ++nt2)
#pragma unroll
      for (int r = 0; r < 8; ++r) {
        int n = mt * 16 + r + half * 8;
        if (n < NTOK) {
          int dd = nt2 * 16 + l15;
          YT[(size_t)(wi * NTOK + n) * DIMC + hd * HDIM + dd] = (bf16)yacc[nt2][r];
        }
      }
  }
}

// ---------------------------------------------------------------------------
// GEMM3: out = Wproj[384x384] * Y ; epilogue undoes window partition + shift
// and writes fp32 to d_out in (B, C, H, W) layout.
// ---------------------------------------------------------------------------
__global__ void __launch_bounds__(256) gemm_proj(const bf16* __restrict__ Wp,
                                                 const bf16* __restrict__ YT,
                                                 float* __restrict__ out) {
  __shared__ __align__(32) bf16 Alds[2048];
  __shared__ __align__(32) bf16 Blds[4096];
  const int row0 = blockIdx.y * 64;
  const int col0 = blockIdx.x * 128;
  const int lane = threadIdx.x & 31;
  const int wave = threadIdx.x >> 5;
  const int wm = wave >> 2, wn = wave & 3;

  v8f acc[2][2] = {{{}, {}}, {{}, {}}};
  gemm_mainloop(Wp, YT, row0, col0, Alds, Blds, acc);

  const int l15 = lane & 15, half = lane >> 4;
  for (int im = 0; im < 2; ++im)
    for (int in = 0; in < 2; ++in)
      for (int r = 0; r < 8; ++r) {
        int o = row0 + wm * 32 + im * 16 + r + half * 8;   // < 384
        int j = col0 + wn * 32 + in * 16 + l15;
        int wi = j / NTOK, n = j - wi * NTOK;
        int b = wi >> 6, g = wi & 63, gy = g >> 3, gx = g & 7;
        int hs = gy * WIN + n / WIN;
        int ws = gx * WIN + (n % WIN);
        int h = hs + 3; if (h >= IMGHW) h -= IMGHW;        // y[(hs+3)%56] = y_shifted[hs]
        int w = ws + 3; if (w >= IMGHW) w -= IMGHW;
        out[(((size_t)b * DIMC + o) * IMGHW + h) * IMGHW + w] = acc[im][in][r];
      }
}

// ---------------------------------------------------------------------------
// Launcher
// ---------------------------------------------------------------------------
extern "C" void kernel_launch(void* const* d_in, const int* in_sizes, int n_in,
                              void* d_out, int out_size, void* d_ws, size_t ws_size,
                              hipStream_t stream) {
  (void)in_sizes; (void)n_in; (void)out_size; (void)ws_size;
  const float* x     = (const float*)d_in[0];
  const float* w_qkv = (const float*)d_in[1];
  const float* w_proj= (const float*)d_in[2];
  const float* cpb   = (const float*)d_in[3];
  float* out = (float*)d_out;

  char* ws = (char*)d_ws;
  size_t off = 0;
  bf16*  XwT  = (bf16*)(ws + off);  off += (size_t)NCOL * DIMC * 2;        // 38,535,168
  bf16*  Wqb  = (bf16*)(ws + off);  off += (size_t)3 * DIMC * DIMC * 2;    //    884,736
  bf16*  Wpb  = (bf16*)(ws + off);  off += (size_t)DIMC * DIMC * 2;        //    294,912
  float* bias = (float*)(ws + off); off += (size_t)HEADS * NPAD * NPAD * 4;//    196,608
  bf16*  Qf   = (bf16*)(ws + off);  off += (size_t)NPAIR * 2048 * 2;       // 50,331,648
  bf16*  Kf   = (bf16*)(ws + off);  off += (size_t)NPAIR * 2048 * 2;
  bf16*  Vf   = (bf16*)(ws + off);  off += (size_t)NPAIR * 2048 * 2;
  bf16*  YT   = (bf16*)(ws + off);  off += (size_t)NCOL * DIMC * 2;

  // 1) weights -> bf16
  cvt_f32_bf16<<<(3 * DIMC * DIMC + 255) / 256, 256, 0, stream>>>(w_qkv, Wqb, 3 * DIMC * DIMC);
  cvt_f32_bf16<<<(DIMC * DIMC + 255) / 256, 256, 0, stream>>>(w_proj, Wpb, DIMC * DIMC);
  // 2) shifted+windowed bf16 input, token-major
  build_xwt<<<(NCOL * DIMC) / 256, 256, 0, stream>>>(x, XwT);
  // 3) dense bias table with padded-column masking
  build_bias<<<(HEADS * NPAD * NPAD) / 256, 256, 0, stream>>>(cpb, bias);
  // 4) zero Q/K/V fragment buffers (padded tokens must be 0)
  {
    int nf4 = (int)((size_t)3 * NPAIR * 2048 * 2 / 16);                    // 9,437,184
    zero_f4<<<(nf4 + 255) / 256, 256, 0, stream>>>((float4*)Qf, nf4);
  }
  // 5) QKV GEMM -> fragment scatter
  gemm_qkv<<<dim3(NCOL / 128, (3 * DIMC) / 64), 256, 0, stream>>>(Wqb, XwT, Qf, Kf, Vf);
  // 6) windowed attention
  attn_kernel<<<NPAIR / 8, 256, 0, stream>>>(Qf, Kf, Vf, bias, YT);
  // 7) output projection + un-window + un-shift
  gemm_proj<<<dim3(NCOL / 128, DIMC / 64), 256, 0, stream>>>(Wpb, YT, out);
}
